// attention_enhance_light_rgb_88527865905545
// MI455X (gfx1250) — matrix operand order repaired and checked by reading, not compiled, not run on hardware
//
#include <hip/hip_runtime.h>
#include <cstdint>
#include <cstddef>

#define USE_ASYNC_LDS 1

typedef _Float16 v16h __attribute__((ext_vector_type(16)));
typedef float    v8f  __attribute__((ext_vector_type(8)));

// ---------- helpers ----------
__device__ __forceinline__ uint32_t pk_f16(float a, float b) {
    union { _Float16 h[2]; uint32_t u; } p;
    p.h[0] = (_Float16)a;
    p.h[1] = (_Float16)b;
    return p.u;
}
__device__ __forceinline__ float f16lo(uint32_t u) {
    union { uint32_t u; _Float16 h[2]; } p; p.u = u; return (float)p.h[0];
}
__device__ __forceinline__ float f16hi(uint32_t u) {
    union { uint32_t u; _Float16 h[2]; } p; p.u = u; return (float)p.h[1];
}

union FragU { uint32_t u[8]; v16h v; };

// A fragment: weights, M = oc (lane&15), K = input channel.
// 16-bit A 16x32 layout: lanes0-15 v0..3 -> K pairs 0..7 (by half), v4..7 -> K=16..31.
__device__ __forceinline__ v16h load_fragA(const uint32_t* sWt, int tap, int ocbase, int lane) {
    FragU f;
    const int m    = ocbase + (lane & 15);
    const int half = (lane >> 4) & 1;
#pragma unroll
    for (int v = 0; v < 8; ++v) {
        const int cp = (v & 3) + 4 * half + 8 * (v >> 2);
        f.u[v] = sWt[(tap * 32 + m) * 16 + cp];
    }
    return f.v;
}

// B fragment: activations, K = input channel, N = pixel (lane&15).
// 16-bit B 32x16 layout: lanes0-15 v0..3 -> K=0..7, v4..7 -> K=16..23; lanes16-31 K=8..15, 24..31.
__device__ __forceinline__ v16h load_fragB(const uint32_t* sAct, int ky, int xbase, int lane) {
    FragU f;
    const int n    = lane & 15;
    const int half = (lane >> 4) & 1;
    const int xl   = xbase + n;
#pragma unroll
    for (int v = 0; v < 8; ++v) {
        const int cp = (v & 3) + 4 * half + 8 * (v >> 2);
        f.u[v] = sAct[(ky * 16 + cp) * 66 + xl];
    }
    return f.v;
}

// ---------- WMMA implicit-GEMM 3x3 conv, 512x512, Cout=32 ----------
// Activations between layers live in packed-f16 channel-pair layout:
//   uint32 at [((b*16 + cp)*512 + y)*512 + x]  holds channels (2cp, 2cp+1).
// F32IN=true (enc1 only): input is f32 NCHW with Cin channels.
// MODE 0: out = relu(conv + bias)
// MODE 1: out = relu(conv - theta[b] * conv1x1(kdiff))        (CDC, no bias)
// MODE 2: out = skip + gamma[b,oc]*(conv + bias) + beta[b,oc] (residual FiLM)
template <int MODE, bool F32IN>
__global__ __launch_bounds__(256) void conv3x3_wmma(
    const void* __restrict__ in, const float* __restrict__ wt,
    const float* __restrict__ bias, uint32_t* __restrict__ out, int Cin,
    const float* __restrict__ theta, const float* __restrict__ gamma,
    const float* __restrict__ beta, const uint32_t* __restrict__ skip)
{
    constexpr int H = 512, W = 512;
    __shared__ uint32_t sAct[3 * 16 * 66];    // [row 0..2][cpair 0..15][x 0..65], f16 pairs
    __shared__ uint32_t sWt[10 * 32 * 16];    // [tap 0..9][oc][cpair], tap9 = kdiff (CDC)

    const int tid = threadIdx.x;
    const int x0  = blockIdx.x * 64;
    const int y   = blockIdx.y;
    const int b   = blockIdx.z;

    // --- stage weights (and kdiff for CDC) as packed f16 channel pairs ---
    constexpr int NTAP = (MODE == 1) ? 10 : 9;
    for (int i = tid; i < NTAP * 512; i += 256) {
        const int tap = i >> 9;
        const int rem = i & 511;
        const int oc  = rem >> 4;
        const int cp  = rem & 15;
        const int c0  = 2 * cp, c1 = 2 * cp + 1;
        float a = 0.f, bv = 0.f;
        if (tap < 9) {
            const int ky = tap / 3, kx = tap - 3 * (tap / 3);
            if (c0 < Cin) a  = wt[((oc * Cin + c0) * 3 + ky) * 3 + kx];
            if (c1 < Cin) bv = wt[((oc * Cin + c1) * 3 + ky) * 3 + kx];
        } else {  // kdiff[oc][c] = sum over 3x3 taps
            if (c0 < Cin) { for (int t = 0; t < 9; ++t) a  += wt[(oc * Cin + c0) * 9 + t]; }
            if (c1 < Cin) { for (int t = 0; t < 9; ++t) bv += wt[(oc * Cin + c1) * 9 + t]; }
        }
        sWt[i] = pk_f16(a, bv);
    }

    // --- stage activation tile: rows y-1..y+1, x0-1..x0+64, zero halo ---
    if constexpr (!F32IN) {
        const uint32_t* inp = (const uint32_t*)in;
        for (int i = tid; i < 3 * 16 * 66; i += 256) {
            const int row = i / (16 * 66);
            const int rem = i - row * (16 * 66);
            const int cp  = rem / 66;
            const int xl  = rem - cp * 66;
            const int yy  = y + row - 1;
            const int gx  = x0 + xl - 1;
            if ((unsigned)yy < (unsigned)H && (unsigned)gx < (unsigned)W) {
                const uint32_t* g = inp + ((size_t)(b * 16 + cp) * H + yy) * W + gx;
#if USE_ASYNC_LDS
                const uint32_t lo = (uint32_t)(uintptr_t)(&sAct[i]);
                asm volatile("global_load_async_to_lds_b32 %0, %1, off"
                             :: "v"(lo), "v"(g) : "memory");
#else
                sAct[i] = *g;
#endif
            } else {
                sAct[i] = 0u;
            }
        }
#if USE_ASYNC_LDS
        asm volatile("s_wait_asynccnt 0x0" ::: "memory");
#endif
    } else {
        const float* inf = (const float*)in;
        for (int i = tid; i < 3 * 16 * 66; i += 256) {
            const int row = i / (16 * 66);
            const int rem = i - row * (16 * 66);
            const int cp  = rem / 66;
            const int xl  = rem - cp * 66;
            const int yy  = y + row - 1;
            const int gx  = x0 + xl - 1;
            const int c0  = 2 * cp, c1 = c0 + 1;
            float a = 0.f, bv = 0.f;
            if ((unsigned)yy < (unsigned)H && (unsigned)gx < (unsigned)W) {
                const float* p = inf + ((size_t)(b * Cin) * H + yy) * W + gx;
                if (c0 < Cin) a  = p[(size_t)c0 * H * W];
                if (c1 < Cin) bv = p[(size_t)c1 * H * W];
            }
            sAct[i] = pk_f16(a, bv);
        }
    }
    __syncthreads();

    // --- per-wave 16oc x 16px tile, one (ky,kx) tap (=32 channels of K) per WMMA ---
    const int lane   = tid & 31;
    const int wv     = tid >> 5;
    const int ocbase = (wv & 1) * 16;
    const int xoff   = (wv >> 1) * 16;

    v8f acc  = {0.f, 0.f, 0.f, 0.f, 0.f, 0.f, 0.f, 0.f};
    v8f accD = {0.f, 0.f, 0.f, 0.f, 0.f, 0.f, 0.f, 0.f};

#pragma unroll
    for (int ky = 0; ky < 3; ++ky) {
#pragma unroll
        for (int kx = 0; kx < 3; ++kx) {
            v16h A  = load_fragA(sWt, ky * 3 + kx, ocbase, lane);
            v16h Bf = load_fragB(sAct, ky, xoff + kx, lane);
            acc = __builtin_amdgcn_wmma_f32_16x16x32_f16(false, A, false, Bf,
                                                         (short)0, acc, false, false);
        }
    }
    if constexpr (MODE == 1) {  // extra 1x1 (kdiff) step on the center tap
        v16h A  = load_fragA(sWt, 9, ocbase, lane);
        v16h Bf = load_fragB(sAct, 1, xoff + 1, lane);
        accD = __builtin_amdgcn_wmma_f32_16x16x32_f16(false, A, false, Bf,
                                                      (short)0, accD, false, false);
    }

    // --- fused epilogue; D layout: M(oc) = r + 8*half, N(px) = lane&15.
    //     Adjacent r hold adjacent oc in the SAME lane -> pack f16 pairs locally. ---
    const int half = lane >> 4;
    const int px   = x0 + xoff + (lane & 15);
    float th = 0.f;
    if constexpr (MODE == 1) th = theta[b];
#pragma unroll
    for (int r = 0; r < 8; r += 2) {
        const int oc0 = ocbase + 8 * half + r;          // even
        const int cp  = oc0 >> 1;
        const size_t oidx = ((size_t)(b * 16 + cp) * H + y) * W + px;
        float v0 = acc[r], v1 = acc[r + 1];
        if constexpr (MODE == 0) {
            v0 = fmaxf(v0 + bias[oc0], 0.f);
            v1 = fmaxf(v1 + bias[oc0 + 1], 0.f);
        } else if constexpr (MODE == 1) {
            v0 = fmaxf(v0 - th * accD[r], 0.f);
            v1 = fmaxf(v1 - th * accD[r + 1], 0.f);
        } else {
            const uint32_t sp = skip[oidx];
            const float cv0 = v0 + bias[oc0];
            const float cv1 = v1 + bias[oc0 + 1];
            v0 = f16lo(sp) + gamma[b * 32 + oc0] * cv0 + beta[b * 32 + oc0];
            v1 = f16hi(sp) + gamma[b * 32 + oc0 + 1] * cv1 + beta[b * 32 + oc0 + 1];
        }
        out[oidx] = pk_f16(v0, v1);
    }
}

// ---------- Haar DWT: (4,3,512,512) -> (4,12,256,256), stride-2 2x2 ----------
__global__ __launch_bounds__(256) void dwt_kernel(const float* __restrict__ x,
                                                  float* __restrict__ dwt) {
    const int id = blockIdx.x * 256 + threadIdx.x;
    if (id >= 4 * 3 * 256 * 256) return;
    const int xo = id & 255;
    const int yo = (id >> 8) & 255;
    const int g  = (id >> 16) % 3;
    const int b  = id / (3 * 65536);
    const float* p = x + ((size_t)(b * 3 + g) * 512 + 2 * yo) * 512 + 2 * xo;
    const float x00 = p[0], x01 = p[1], x10 = p[512], x11 = p[513];
    float* o = dwt + ((size_t)(b * 12 + 4 * g) * 256 + yo) * 256 + xo;
    const size_t cs = 256 * 256;
    o[0]      = 0.5f * ( x00 + x01 + x10 + x11);   // LL
    o[cs]     = 0.5f * ( x00 + x01 - x10 - x11);   // LH (pre-flipped)
    o[2 * cs] = 0.5f * ( x00 - x01 + x10 - x11);   // HL
    o[3 * cs] = 0.5f * ( x00 - x01 - x10 + x11);   // HH
}

// ---------- ct branch: conv3x3(high_freq 9ch -> 64) + relu, spatial sum ----------
__global__ __launch_bounds__(256) void ct_reduce_kernel(const float* __restrict__ dwt,
                                                        const float* __restrict__ ct_w,
                                                        const float* __restrict__ ct_b,
                                                        float* __restrict__ ct_sum) {
    const int oc = blockIdx.x, b = blockIdx.y, tid = threadIdx.x;
    __shared__ float w[81];
    __shared__ float red[256];
    if (tid < 81) w[tid] = ct_w[oc * 81 + tid];
    __syncthreads();
    const float bia = ct_b[oc];
    float s = 0.f;
    for (int pix = tid; pix < 65536; pix += 256) {
        const int yo = pix >> 8, xo = pix & 255;
        float v = bia;
        for (int j = 0; j < 9; ++j) {
            const int ch = (j % 3) * 4 + (j / 3) + 1;   // high_freq channel map
            const float* p = dwt + (size_t)(b * 12 + ch) * 65536;
            for (int dy = 0; dy < 3; ++dy) {
                const int yy = yo + dy - 1;
                if ((unsigned)yy >= 256u) continue;
                for (int dx = 0; dx < 3; ++dx) {
                    const int xx = xo + dx - 1;
                    if ((unsigned)xx >= 256u) continue;
                    v += p[yy * 256 + xx] * w[j * 9 + dy * 3 + dx];
                }
            }
        }
        s += fmaxf(v, 0.f);
    }
    red[tid] = s;
    __syncthreads();
    for (int st = 128; st > 0; st >>= 1) {
        if (tid < st) red[tid] += red[tid + st];
        __syncthreads();
    }
    if (tid == 0) ct_sum[b * 64 + oc] = red[0];
}

// ---------- lf branch: conv3x3(haar_A 3ch -> 64) + relu, spatial sum ----------
__global__ __launch_bounds__(256) void lf_reduce_kernel(const float* __restrict__ dwt,
                                                        const float* __restrict__ lf_w,
                                                        const float* __restrict__ lf_b,
                                                        float* __restrict__ lf_sum) {
    const int oc = blockIdx.x, b = blockIdx.y, tid = threadIdx.x;
    __shared__ float w[27];
    __shared__ float red[256];
    if (tid < 27) w[tid] = lf_w[oc * 27 + tid];
    __syncthreads();
    const float bia = lf_b[oc];
    float s = 0.f;
    for (int pix = tid; pix < 65536; pix += 256) {
        const int yo = pix >> 8, xo = pix & 255;
        float v = bia;
        for (int g = 0; g < 3; ++g) {
            const float* p = dwt + (size_t)(b * 12 + 4 * g) * 65536;  // haar_A ch 0,4,8
            for (int dy = 0; dy < 3; ++dy) {
                const int yy = yo + dy - 1;
                if ((unsigned)yy >= 256u) continue;
                for (int dx = 0; dx < 3; ++dx) {
                    const int xx = xo + dx - 1;
                    if ((unsigned)xx >= 256u) continue;
                    v += p[yy * 256 + xx] * w[g * 9 + dy * 3 + dx];
                }
            }
        }
        s += fmaxf(v, 0.f);
    }
    red[tid] = s;
    __syncthreads();
    for (int st = 128; st > 0; st >>= 1) {
        if (tid < st) red[tid] += red[tid + st];
        __syncthreads();
    }
    if (tid == 0) lf_sum[b * 64 + oc] = red[0];
}

// ---------- tiny head: theta / gamma / beta from pooled sums (1 block) ----------
__global__ __launch_bounds__(512) void head_kernel(
    const float* __restrict__ ct_sum, const float* __restrict__ lf_sum,
    const float* __restrict__ t1_w, const float* __restrict__ t1_b,
    const float* __restrict__ t2_w, const float* __restrict__ t2_b,
    const float* __restrict__ t3_w, const float* __restrict__ t3_b,
    const float* __restrict__ fc1_w, const float* __restrict__ fc1_b,
    const float* __restrict__ fc2_w, const float* __restrict__ fc2_b,
    const float* __restrict__ fc3_w, const float* __restrict__ fc3_b,
    const float* __restrict__ g_w, const float* __restrict__ g_b,
    const float* __restrict__ be_w, const float* __restrict__ be_b,
    float* __restrict__ theta, float* __restrict__ gamma, float* __restrict__ beta)
{
    __shared__ float bufA[512];
    __shared__ float bufB[512];
    const int t = threadIdx.x;
    const float inv = 1.f / 65536.f;
    for (int b = 0; b < 4; ++b) {
        // theta path
        if (t < 64) bufA[t] = ct_sum[b * 64 + t] * inv;
        __syncthreads();
        if (t < 64) {
            float s = t1_b[t];
            for (int c = 0; c < 64; ++c) s += t1_w[t * 64 + c] * bufA[c];
            bufB[t] = fmaxf(s, 0.f);
        }
        __syncthreads();
        if (t < 64) {
            float s = t2_b[t];
            for (int c = 0; c < 64; ++c) s += t2_w[t * 64 + c] * bufB[c];
            bufA[t] = fmaxf(s, 0.f);
        }
        __syncthreads();
        if (t == 0) {
            float s = t3_b[0];
            for (int c = 0; c < 64; ++c) s += t3_w[c] * bufA[c];
            theta[b] = 1.f / (1.f + expf(-s));
        }
        __syncthreads();
        // am path
        if (t < 64) bufA[t] = lf_sum[b * 64 + t] * inv;
        __syncthreads();
        {
            float s = fc1_b[t];
            for (int c = 0; c < 64; ++c) s += fc1_w[t * 64 + c] * bufA[c];
            bufB[t] = fmaxf(s, 0.f);
        }
        __syncthreads();
        {
            float s = fc2_b[t];
            for (int c = 0; c < 512; ++c) s += fc2_w[t * 512 + c] * bufB[c];
            bufA[t] = fmaxf(s, 0.f);
        }
        __syncthreads();
        {
            float s = fc3_b[t];
            for (int c = 0; c < 512; ++c) s += fc3_w[t * 512 + c] * bufA[c];
            bufB[t] = 1.f / (1.f + expf(-s));
        }
        __syncthreads();
        if (t < 32) {
            float s1 = g_b[t], s2 = be_b[t];
            for (int c = 0; c < 512; ++c) {
                s1 += g_w[t * 512 + c] * bufB[c];
                s2 += be_w[t * 512 + c] * bufB[c];
            }
            gamma[b * 32 + t] = 1.f / (1.f + expf(-s1));
            beta[b * 32 + t]  = tanhf(s2);
        }
        __syncthreads();
    }
}

// ---------- d2 conv (32->1, packed-f16 input) + tanh + enhancement ----------
__global__ __launch_bounds__(256) void d2_enhance_kernel(const uint32_t* __restrict__ f,
                                                         const float* __restrict__ x,
                                                         const float* __restrict__ d2_w,
                                                         const float* __restrict__ d2_b,
                                                         float* __restrict__ out) {
    const int id = blockIdx.x * 256 + threadIdx.x;
    if (id >= 4 * 512 * 512) return;
    const int xx = id & 511;
    const int yy = (id >> 9) & 511;
    const int b  = id >> 18;
    float s = d2_b[0];
    for (int cp = 0; cp < 16; ++cp) {
        const uint32_t* p = f + (size_t)(b * 16 + cp) * 512 * 512;
        const float* w0 = d2_w + (2 * cp) * 9;
        const float* w1 = d2_w + (2 * cp + 1) * 9;
        for (int dy = 0; dy < 3; ++dy) {
            const int y2 = yy + dy - 1;
            if ((unsigned)y2 >= 512u) continue;
            for (int dx = 0; dx < 3; ++dx) {
                const int x2 = xx + dx - 1;
                if ((unsigned)x2 >= 512u) continue;
                const uint32_t u = p[y2 * 512 + x2];
                s += f16lo(u) * w0[dy * 3 + dx] + f16hi(u) * w1[dy * 3 + dx];
            }
        }
    }
    const float t = tanhf(s);
    const size_t HW = 512 * 512;
    for (int g = 0; g < 3; ++g) {
        const size_t idx = (size_t)(b * 3 + g) * HW + (size_t)yy * 512 + xx;
        const float xv = x[idx];
        out[idx] = xv + t * (xv * xv - xv);
    }
    out[12 * HW + (size_t)b * HW + (size_t)yy * 512 + xx] = t;
}

// ---------- launch ----------
extern "C" void kernel_launch(void* const* d_in, const int* in_sizes, int n_in,
                              void* d_out, int out_size, void* d_ws, size_t ws_size,
                              hipStream_t stream) {
    const float* x      = (const float*)d_in[0];
    const float* enc1_w = (const float*)d_in[1];  const float* enc1_b = (const float*)d_in[2];
    const float* enc2_w = (const float*)d_in[3];  const float* enc2_b = (const float*)d_in[4];
    const float* enc3_w = (const float*)d_in[5];  const float* enc3_b = (const float*)d_in[6];
    const float* enc4_w = (const float*)d_in[7];  const float* enc4_b = (const float*)d_in[8];
    const float* enc5_w = (const float*)d_in[9];  const float* enc5_b = (const float*)d_in[10];
    const float* cdc_w  = (const float*)d_in[11];
    const float* ct_w   = (const float*)d_in[12]; const float* ct_b   = (const float*)d_in[13];
    const float* t1_w   = (const float*)d_in[14]; const float* t1_b   = (const float*)d_in[15];
    const float* t2_w   = (const float*)d_in[16]; const float* t2_b   = (const float*)d_in[17];
    const float* t3_w   = (const float*)d_in[18]; const float* t3_b   = (const float*)d_in[19];
    const float* lf_w   = (const float*)d_in[20]; const float* lf_b   = (const float*)d_in[21];
    const float* fc1_w  = (const float*)d_in[22]; const float* fc1_b  = (const float*)d_in[23];
    const float* fc2_w  = (const float*)d_in[24]; const float* fc2_b  = (const float*)d_in[25];
    const float* fc3_w  = (const float*)d_in[26]; const float* fc3_b  = (const float*)d_in[27];
    const float* g_w    = (const float*)d_in[28]; const float* g_b    = (const float*)d_in[29];
    const float* be_w   = (const float*)d_in[30]; const float* be_b   = (const float*)d_in[31];
    const float* res_w  = (const float*)d_in[32]; const float* res_b  = (const float*)d_in[33];
    const float* d1_w   = (const float*)d_in[34]; const float* d1_b   = (const float*)d_in[35];
    const float* d2_w   = (const float*)d_in[36]; const float* d2_b   = (const float*)d_in[37];

    char* ws = (char*)d_ws;
    const size_t FEAT = (size_t)4 * 16 * 512 * 512 * sizeof(uint32_t);  // 67 MB packed f16
    uint32_t* buf0 = (uint32_t*)(ws);
    uint32_t* buf1 = (uint32_t*)(ws + FEAT);
    float* dwtb    = (float*)(ws + 2 * FEAT);                           // 12.6 MB
    float* ct_sum  = (float*)(ws + 2 * FEAT + (size_t)4 * 12 * 256 * 256 * sizeof(float));
    float* lf_sum  = ct_sum + 256;
    float* theta   = lf_sum + 256;
    float* gamma   = theta + 64;
    float* beta    = gamma + 128;

    const dim3 cgrid(512 / 64, 512, 4), cblk(256);

    // encoder chain (ping-pong): f_org4 ends in buf0
    conv3x3_wmma<0, true ><<<cgrid, cblk, 0, stream>>>(x,    enc1_w, enc1_b, buf0, 3,  nullptr, nullptr, nullptr, nullptr);
    conv3x3_wmma<0, false><<<cgrid, cblk, 0, stream>>>(buf0, enc2_w, enc2_b, buf1, 32, nullptr, nullptr, nullptr, nullptr);
    conv3x3_wmma<0, false><<<cgrid, cblk, 0, stream>>>(buf1, enc3_w, enc3_b, buf0, 32, nullptr, nullptr, nullptr, nullptr);
    conv3x3_wmma<0, false><<<cgrid, cblk, 0, stream>>>(buf0, enc4_w, enc4_b, buf1, 32, nullptr, nullptr, nullptr, nullptr);
    conv3x3_wmma<0, false><<<cgrid, cblk, 0, stream>>>(buf1, enc5_w, enc5_b, buf0, 32, nullptr, nullptr, nullptr, nullptr);

    // attention branch (tiny)
    dwt_kernel<<<(4 * 3 * 256 * 256) / 256, 256, 0, stream>>>(x, dwtb);
    ct_reduce_kernel<<<dim3(64, 4), 256, 0, stream>>>(dwtb, ct_w, ct_b, ct_sum);
    lf_reduce_kernel<<<dim3(64, 4), 256, 0, stream>>>(dwtb, lf_w, lf_b, lf_sum);
    head_kernel<<<1, 512, 0, stream>>>(ct_sum, lf_sum, t1_w, t1_b, t2_w, t2_b, t3_w, t3_b,
                                       fc1_w, fc1_b, fc2_w, fc2_b, fc3_w, fc3_b,
                                       g_w, g_b, be_w, be_b, theta, gamma, beta);

    // CDC: f_deco = relu(conv(f_org4) - theta * kdiff_conv(f_org4))  -> buf1
    conv3x3_wmma<1, false><<<cgrid, cblk, 0, stream>>>(buf0, cdc_w, nullptr, buf1, 32, theta, nullptr, nullptr, nullptr);
    // residual FiLM: f = f_deco + gamma*(conv(f_deco)+bias) + beta   -> buf0
    conv3x3_wmma<2, false><<<cgrid, cblk, 0, stream>>>(buf1, res_w, res_b, buf0, 32, nullptr, gamma, beta, buf1);
    // d1: relu conv -> buf1
    conv3x3_wmma<0, false><<<cgrid, cblk, 0, stream>>>(buf0, d1_w, d1_b, buf1, 32, nullptr, nullptr, nullptr, nullptr);

    // d2 + tanh + enhancement, writes (enhance_image, out_map) into d_out
    d2_enhance_kernel<<<(4 * 512 * 512) / 256, 256, 0, stream>>>(buf1, x, d2_w, d2_b, (float*)d_out);
}
